// RNNCell_24429773979888
// MI455X (gfx1250) — matrix operand order repaired and checked by reading, not compiled, not run on hardware
//
#include <hip/hip_runtime.h>
#include <hip/hip_bf16.h>

// ---------------------------------------------------------------------------
// MI455X (gfx1250, wave32) fused STSP-RNN cell.
//   Pass 1: elementwise synapse update -> syn_x/syn_u outputs + h_post (bf16)
//   Pass 2: weight prep: wrnn = relu(rnnmat)*ei -> bf16
//   Pass 3: small prep: inputs->bf16, Bcat = [Wmd_a | Win] (K=288),
//           Bx = [Wmd_x | 0] (K=32, zero-padded for WMMA)
//   Pass 4: fused GEMM (v_wmma_f32_16x16x32_bf16):
//           rec (K=2048), aux=inp+md_a (K=288), md_x (K=32 padded)
//           + tanh/relu/leaky-integrate epilogue -> h
// Addressing in pass 4 uses uniform-base + 32-bit lane offsets so the backend
// can pick SADDR-form global loads (loop-invariant voffset, scalar k advance).
// ---------------------------------------------------------------------------

typedef __bf16 bf16_t;
typedef __attribute__((ext_vector_type(16))) __bf16 v16bf;
typedef __attribute__((ext_vector_type(8)))  __bf16 v8bf;
typedef __attribute__((ext_vector_type(8)))  float  v8f;

#define NB    1024      // batch
#define NH    2048      // hidden
#define NIN   288       // 16 rule + 256 tuned
#define NEXC  1638      // int(0.8 * 2048)
#define ALPHA_F 0.1f            // DT/TAU = 10/100
#define NOISE_C 0.223606797749979f  // NOISE_SD * sqrt(2*ALPHA) = 0.5*sqrt(0.2)

static __device__ __forceinline__ v8f wmma_bf16(v16bf a, v16bf b, v8f c) {
    return __builtin_amdgcn_wmma_f32_16x16x32_bf16(
        /*neg_a=*/false, a, /*neg_b=*/false, b,
        /*c_mod=*/(short)0, c, /*reuse_a=*/false, /*reuse_b=*/false);
}

// Load a 16-half WMMA fragment as two contiguous 16-byte chunks.
// A fragment (16x32) and B fragment (32x16): both chunk1 at +8 elems (contiguous k).
static __device__ __forceinline__ v16bf ld_frag(const bf16_t* __restrict__ p,
                                                int second_off) {
    union { v16bf v; v8bf h[2]; } u;
    u.h[0] = *reinterpret_cast<const v8bf*>(p);
    u.h[1] = *reinterpret_cast<const v8bf*>(p + second_off);
    return u.v;
}

// ---------------- Pass 1: synapse update + h_post (bf16) -------------------
__global__ void k_prep_syn(const float* __restrict__ prev_h,
                           const float* __restrict__ syn_x_in,
                           const float* __restrict__ syn_u_in,
                           float* __restrict__ out_syn_x,
                           float* __restrict__ out_syn_u,
                           bf16_t* __restrict__ hpost_bf) {
    int idx = blockIdx.x * blockDim.x + threadIdx.x;
    if (idx >= NB * NH) return;
    int j = idx & (NH - 1);
    bool even = (j & 1) == 0;
    float a_std = even ? 0.05f : 0.00417f;
    float px = prev_h[idx], sx = syn_x_in[idx], su = syn_u_in[idx];
    float sxn = sx + a_std * (1.0f - sx) - 0.01f * su * sx * px;  // DT/1000 = 0.01
    sxn = fminf(1.0f, fmaxf(0.0f, sxn));
    // reference overwrites syn_u with clamp(relu(syn_x)) -> equals sxn:
    float sun = fminf(1.0f, fmaxf(0.0f, sxn));
    out_syn_x[idx] = sxn;
    out_syn_u[idx] = sun;
    hpost_bf[idx] = (bf16_t)(sun * sxn * px);
}

// ---------------- Pass 2: wrnn = relu(rnnmat) * ei  -> bf16 ----------------
__global__ void k_prep_w(const float* __restrict__ rnnmat,
                         bf16_t* __restrict__ wrnn_bf) {
    int idx = blockIdx.x * blockDim.x + threadIdx.x;
    if (idx >= NH * NH) return;
    int j = idx & (NH - 1);
    float w = fmaxf(rnnmat[idx], 0.0f);
    wrnn_bf[idx] = (bf16_t)(j < NEXC ? w : -w);
}

// ---------------- Pass 3: small conversions --------------------------------
__global__ void k_prep_small(const float* __restrict__ inputs,
                             const float* __restrict__ Win,
                             const float* __restrict__ Wmd_x,
                             const float* __restrict__ Wmd_a,
                             bf16_t* __restrict__ in_bf,     // [NB][NIN]
                             bf16_t* __restrict__ bcat_bf,   // [NH][NIN] = [Wmd_a|Win]
                             bf16_t* __restrict__ bx_bf) {   // [NH][32]  = [Wmd_x|0]
    int t = blockIdx.x * blockDim.x + threadIdx.x;
    const int N_IN = NB * NIN;   // 294912
    const int N_BC = NH * NIN;   // 589824
    const int N_BX = NH * 32;    // 65536
    if (t < N_IN) { in_bf[t] = (bf16_t)inputs[t]; return; }
    t -= N_IN;
    if (t < N_BC) {
        int i = t / NIN, k = t - i * NIN;
        float v = (k < 16) ? Wmd_a[i * 16 + k] : Win[i * 256 + (k - 16)];
        bcat_bf[t] = (bf16_t)v;
        return;
    }
    t -= N_BC;
    if (t < N_BX) {
        int i = t >> 5, k = t & 31;
        bx_bf[t] = (bf16_t)((k < 16) ? Wmd_x[i * 16 + k] : 0.0f);
    }
}

// ---------------- Pass 4: fused WMMA GEMM + epilogue -----------------------
// Block: 128 threads = 4 wave32s; block tile 64x64; wave tile 32x32 (2x2 WMMA).
__global__ __launch_bounds__(128)
void k_rnn_gemm(const bf16_t* __restrict__ Ah,    // h_post bf16 [NB][NH]
                const bf16_t* __restrict__ Bw,    // wrnn  bf16 [NH][NH]
                const bf16_t* __restrict__ Ain,   // inputs bf16 [NB][NIN]
                const bf16_t* __restrict__ Bcat,  // [NH][NIN]
                const bf16_t* __restrict__ Bx,    // [NH][32]
                const float* __restrict__ noise,
                const float* __restrict__ prev_h,
                float* __restrict__ out_h) {
    const int lane = threadIdx.x & 31;
    const int wave = threadIdx.x >> 5;
    const int l  = lane & 15;
    const int hi = lane >> 4;
    const int m0 = blockIdx.y * 64 + (wave >> 1) * 32;  // batch rows
    const int n0 = blockIdx.x * 64 + (wave & 1) * 32;   // hidden cols

    // Loop-invariant 32-bit per-lane element offsets (uniform base + offset + k).
    const int aOffH0 = (m0 + l) * NH + hi * 16;       // A frag, M tile 0
    const int aOffH1 = aOffH0 + 16 * NH;              // A frag, M tile 1
    const int bOffW0 = (n0 + l) * NH + hi * 16;       // B frag, N tile 0
    const int bOffW1 = bOffW0 + 16 * NH;              // B frag, N tile 1
    const int aOffI0 = (m0 + l) * NIN + hi * 16;
    const int aOffI1 = aOffI0 + 16 * NIN;
    const int bOffC0 = (n0 + l) * NIN + hi * 16;
    const int bOffC1 = bOffC0 + 16 * NIN;
    const int bOffX0 = (n0 + l) * 32 + hi * 16;
    const int bOffX1 = bOffX0 + 16 * 32;

    v8f accR[2][2], accA[2][2], accX[2][2];
#pragma unroll
    for (int i = 0; i < 2; ++i)
#pragma unroll
        for (int j = 0; j < 2; ++j) {
            accR[i][j] = (v8f)0.f; accA[i][j] = (v8f)0.f; accX[i][j] = (v8f)0.f;
        }

    // ---- rec = h_post @ wrnn.T, K = 2048 ----
    for (int k = 0; k < NH; k += 32) {
        // Unconditional prefetch of the streaming W operand 8 k-steps ahead;
        // overshoot past wrnn's end lands in mapped workspace (speculative OK).
        __builtin_prefetch(Bw + bOffW0 + k + 256, 0, 1);
        __builtin_prefetch(Bw + bOffW1 + k + 256, 0, 1);
        v16bf a0 = ld_frag(Ah + aOffH0 + k, 8);
        v16bf a1 = ld_frag(Ah + aOffH1 + k, 8);
        v16bf b0 = ld_frag(Bw + bOffW0 + k, 8);
        v16bf b1 = ld_frag(Bw + bOffW1 + k, 8);
        accR[0][0] = wmma_bf16(a0, b0, accR[0][0]);
        accR[0][1] = wmma_bf16(a0, b1, accR[0][1]);
        accR[1][0] = wmma_bf16(a1, b0, accR[1][0]);
        accR[1][1] = wmma_bf16(a1, b1, accR[1][1]);
    }

    // ---- aux = inputs @ [Wmd_a|Win].T, K = 288 ----
#pragma unroll
    for (int k = 0; k < NIN; k += 32) {
        v16bf a0 = ld_frag(Ain + aOffI0 + k, 8);
        v16bf a1 = ld_frag(Ain + aOffI1 + k, 8);
        v16bf b0 = ld_frag(Bcat + bOffC0 + k, 8);
        v16bf b1 = ld_frag(Bcat + bOffC1 + k, 8);
        accA[0][0] = wmma_bf16(a0, b0, accA[0][0]);
        accA[0][1] = wmma_bf16(a0, b1, accA[0][1]);
        accA[1][0] = wmma_bf16(a1, b0, accA[1][0]);
        accA[1][1] = wmma_bf16(a1, b1, accA[1][1]);
    }

    // ---- md_x = rule @ Wmd_x.T, K = 16 zero-padded to 32 ----
    {
        v16bf a0 = ld_frag(Ain + aOffI0, 8);
        v16bf a1 = ld_frag(Ain + aOffI1, 8);
        v16bf b0 = ld_frag(Bx + bOffX0, 8);
        v16bf b1 = ld_frag(Bx + bOffX1, 8);
        accX[0][0] = wmma_bf16(a0, b0, accX[0][0]);
        accX[0][1] = wmma_bf16(a0, b1, accX[0][1]);
        accX[1][0] = wmma_bf16(a1, b0, accX[1][0]);
        accX[1][1] = wmma_bf16(a1, b1, accX[1][1]);
    }

    // ---- epilogue: dh = (1+md_x)*rec + aux + c*noise; h = leaky tanh ----
#pragma unroll
    for (int tm = 0; tm < 2; ++tm)
#pragma unroll
        for (int tn = 0; tn < 2; ++tn) {
            union { v8f v; float f[8]; } uR, uA, uX;
            uR.v = accR[tm][tn]; uA.v = accA[tm][tn]; uX.v = accX[tm][tn];
            const int n = n0 + tn * 16 + l;
#pragma unroll
            for (int r = 0; r < 8; ++r) {
                const int m = m0 + tm * 16 + hi * 8 + r;
                const int idx = m * NH + n;
                float dh = (1.0f + uX.f[r]) * uR.f[r] + uA.f[r] + NOISE_C * noise[idx];
                float t = tanhf(dh);
                t = t > 0.0f ? t : 0.0f;
                out_h[idx] = (1.0f - ALPHA_F) * prev_h[idx] + ALPHA_F * t;
            }
        }
}

// ---------------------------------------------------------------------------
extern "C" void kernel_launch(void* const* d_in, const int* in_sizes, int n_in,
                              void* d_out, int out_size, void* d_ws, size_t ws_size,
                              hipStream_t stream) {
    (void)in_sizes; (void)n_in; (void)out_size; (void)ws_size;
    const float* inputs = (const float*)d_in[0];  // [NB][NIN]
    const float* prev_h = (const float*)d_in[1];  // [NB][NH]
    const float* syn_x  = (const float*)d_in[2];
    const float* syn_u  = (const float*)d_in[3];
    const float* noise  = (const float*)d_in[4];
    const float* rnnmat = (const float*)d_in[5];  // [NH][NH]
    const float* Win    = (const float*)d_in[6];  // [NH][256]
    const float* Wmd_x  = (const float*)d_in[7];  // [NH][16]
    const float* Wmd_a  = (const float*)d_in[8];  // [NH][16]

    float* out = (float*)d_out;
    float* out_h  = out;                       // [NB][NH]
    float* out_sx = out + (size_t)NB * NH;     // [NB][NH]
    float* out_su = out + (size_t)2 * NB * NH; // [NB][NH]

    char* ws = (char*)d_ws;
    bf16_t* hpost_bf = (bf16_t*)(ws);                          //  4 MB
    bf16_t* wrnn_bf  = (bf16_t*)(ws + ((size_t)4  << 20));     //  8 MB
    bf16_t* in_bf    = (bf16_t*)(ws + ((size_t)12 << 20));     // 576 KB
    bf16_t* bcat_bf  = (bf16_t*)(ws + ((size_t)13 << 20));     // ~1.13 MB
    bf16_t* bx_bf    = (bf16_t*)(ws + ((size_t)15 << 20));     // 128 KB

    // Pass 1: synapse update + h_post
    k_prep_syn<<<(NB * NH + 255) / 256, 256, 0, stream>>>(
        prev_h, syn_x, syn_u, out_sx, out_su, hpost_bf);
    // Pass 2: recurrent weight prep
    k_prep_w<<<(NH * NH + 255) / 256, 256, 0, stream>>>(rnnmat, wrnn_bf);
    // Pass 3: small conversions
    {
        int total = NB * NIN + NH * NIN + NH * 32;
        k_prep_small<<<(total + 255) / 256, 256, 0, stream>>>(
            inputs, Win, Wmd_x, Wmd_a, in_bf, bcat_bf, bx_bf);
    }
    // Pass 4: fused WMMA GEMM + epilogue
    dim3 grid(NH / 64, NB / 64);   // (32, 16)
    k_rnn_gemm<<<grid, 128, 0, stream>>>(
        hpost_bf, wrnn_bf, in_bf, bcat_bf, bx_bf, noise, prev_h, out_h);
}
